// ICON_Transformer_89885075571005
// MI455X (gfx1250) — matrix operand order repaired and checked by reading, not compiled, not run on hardware
//
#include <hip/hip_runtime.h>

typedef __attribute__((ext_vector_type(16))) _Float16 v16h;
typedef __attribute__((ext_vector_type(8)))  float    v8f;

#define R_TOTAL 131072   // B*N*NH rows
#define EDIM    256
#define FFDIM   1024
#define RPB     64       // rows per workgroup (4 row-tiles of 16)
#define NTHR    256      // 8 wave32

__device__ __forceinline__ float silu_f(float x) { return x / (1.0f + __expf(-x)); }

__device__ __forceinline__ v8f wmma_f16(v16h a, v16h b, v8f c) {
  return __builtin_amdgcn_wmma_f32_16x16x32_f16(false, a, false, b, (short)0, c, false, false);
}

// A-fragment LDS slot (dword index) for element pair (row 0..63, even col c0 of 256).
// Layout per ISA 16-bit A 16x32: lane = half*16 + m ; vgpr = blk*4 + g ;
// K pair base = blk*16 + half*8 + g*2 within the 32-wide K tile.
__device__ __forceinline__ int fragAddrA(int row, int c0) {
  int rt = row >> 4, m = row & 15;
  int kt = c0 >> 5, rem = c0 & 31;
  int blk = rem >> 4, rem2 = rem & 15;
  int half = rem2 >> 3, g = (rem2 & 7) >> 1;
  return (((rt * 8 + kt) * 32) + (half * 16 + m)) * 8 + (blk * 4 + g);
}

__device__ __forceinline__ v16h loadAfrag(const unsigned int* Af, int rt, int kt, int lane) {
  return *(const v16h*)(const void*)(Af + ((((rt * 8 + kt) * 32) + lane) << 3));
}

// B fragment: packed weight buffer, one contiguous 32B chunk per (kt,nt,lane).
__device__ __forceinline__ v16h loadBfrag(const unsigned int* pk, int NT, int kt, int nt, int lane) {
  return *(const v16h*)(const void*)(pk + ((((size_t)kt * NT + nt) * 32 + lane) << 3));
}

// A fragment gathered from plain row-major f16 LDS (stride must be even).
__device__ __forceinline__ v16h loadAfragPlain(const _Float16* base, int stride, int rt, int kt, int lane) {
  int m = lane & 15, half = lane >> 4;
  const _Float16* rowp = base + (rt * 16 + m) * stride;
  union { v16h v; unsigned int u[8]; } r;
#pragma unroll
  for (int vg = 0; vg < 8; ++vg) {
    int blk = vg >> 2, g = vg & 3;
    int k0 = kt * 32 + blk * 16 + half * 8 + g * 2;
    r.u[vg] = *(const unsigned int*)(const void*)(rowp + k0);
  }
  return r.v;
}

// GEMM over one n-tile, all 4 row-tiles: B fragment loaded once per kt,
// reused 4x (cuts L2 weight traffic 4x vs per-(rt,nt) jobs).
template <int KT>
__device__ __forceinline__ void gemm4(const unsigned int* Af, const unsigned int* pk,
                                      int NT, int nt, int lane, v8f acc[4]) {
#pragma unroll
  for (int kt = 0; kt < KT; ++kt) {
    v16h b = loadBfrag(pk, NT, kt, nt, lane);
#pragma unroll
    for (int rt = 0; rt < 4; ++rt)
      acc[rt] = wmma_f16(loadAfrag(Af, rt, kt, lane), b, acc[rt]);
  }
}

template <int KT>
__device__ __forceinline__ void gemm4_plain(const _Float16* A, int stride, const unsigned int* pk,
                                            int NT, int nt, int lane, v8f acc[4]) {
#pragma unroll
  for (int kt = 0; kt < KT; ++kt) {
    v16h b = loadBfrag(pk, NT, kt, nt, lane);
#pragma unroll
    for (int rt = 0; rt < 4; ++rt)
      acc[rt] = wmma_f16(loadAfragPlain(A, stride, rt, kt, lane), b, acc[rt]);
  }
}

// ---------------- weight pack: fp32 (K x N) -> f16 B-fragments ----------------
__global__ void pack_weight_kernel(const float* __restrict__ W, unsigned int* __restrict__ out,
                                   int K, int N) {
  int i = blockIdx.x * blockDim.x + threadIdx.x;        // dword index
  int total = (K >> 5) * (N >> 4) * 256;
  if (i >= total) return;
  int vg = i & 7, lane = (i >> 3) & 31, frag = i >> 8;
  int NT = N >> 4;
  int kt = frag / NT, nt = frag - kt * NT;
  int n = nt * 16 + (lane & 15);
  int k0 = kt * 32 + (vg >> 2) * 16 + (lane >> 4) * 8 + (vg & 3) * 2;
  union { _Float16 h[2]; unsigned int u; } cv;
  cv.h[0] = (_Float16)W[(size_t)k0 * N + n];
  cv.h[1] = (_Float16)W[(size_t)(k0 + 1) * N + n];
  out[i] = cv.u;
}

// ---------------- positional bias: silu([p1,p2]@Wpos)@Wbias ----------------
__global__ void posbias_kernel(const float* __restrict__ p1, const float* __restrict__ p2,
                               const float* __restrict__ Wp, const float* __restrict__ Wb,
                               float* __restrict__ pb) {
  int r = blockIdx.x * blockDim.x + threadIdx.x;
  if (r >= R_TOTAL) return;
  float pv0 = p1[2 * r], pv1 = p1[2 * r + 1], pv2 = p2[2 * r], pv3 = p2[2 * r + 1];
  float acc[8] = {0.f, 0.f, 0.f, 0.f, 0.f, 0.f, 0.f, 0.f};
  for (int j = 0; j < 32; ++j) {
    float e = pv0 * Wp[j] + pv1 * Wp[32 + j] + pv2 * Wp[64 + j] + pv3 * Wp[96 + j];
    e = silu_f(e);
#pragma unroll
    for (int h = 0; h < 8; ++h) acc[h] += e * Wb[j * 8 + h];
  }
#pragma unroll
  for (int h = 0; h < 8; ++h) pb[(size_t)r * 8 + h] = acc[h];
}

// ---------------- kernel 1: t = LN(x @ W_in) ----------------
__global__ __launch_bounds__(NTHR, 1)
void inproj_ln_kernel(const float* __restrict__ x, const unsigned int* __restrict__ pkW,
                      const float* __restrict__ gv, const float* __restrict__ bvv,
                      _Float16* __restrict__ t16) {
  __shared__ alignas(32) unsigned int Af[4 * 8 * 32 * 8];   // 32 KB: A fragments (64x256 f16)
  __shared__ float ys[RPB * 260];                           // fp32 result tile
  __shared__ float redA[256], redB[256];
  int tid = threadIdx.x, lane = tid & 31, wave = tid >> 5;
  size_t r0 = (size_t)blockIdx.x * RPB;

  const float2* src = (const float2*)(x + r0 * EDIM);
  for (int it = 0; it < 32; ++it) {
    int idx = it * NTHR + tid;
    int row = idx >> 7, pr = idx & 127;
    float2 f = src[row * 128 + pr];
    union { _Float16 h[2]; unsigned int u; } cv;
    cv.h[0] = (_Float16)f.x; cv.h[1] = (_Float16)f.y;
    Af[fragAddrA(row, pr * 2)] = cv.u;
  }
  __syncthreads();

  for (int nt = wave; nt < 16; nt += 8) {
    v8f acc[4] = {};
    gemm4<8>(Af, pkW, 16, nt, lane, acc);
    int col = nt * 16 + (lane & 15);
    int mbase = (lane >> 4) * 8;
#pragma unroll
    for (int rt = 0; rt < 4; ++rt)
#pragma unroll
      for (int i = 0; i < 8; ++i) ys[(rt * 16 + mbase + i) * 260 + col] = acc[rt][i];
  }
  __syncthreads();

  int row = tid >> 2, part = tid & 3;
  float s = 0.f, s2 = 0.f;
  for (int c = part * 64; c < part * 64 + 64; ++c) { float v = ys[row * 260 + c]; s += v; s2 += v * v; }
  redA[tid] = s; redB[tid] = s2;
  __syncthreads();
  float mean = (redA[row * 4] + redA[row * 4 + 1] + redA[row * 4 + 2] + redA[row * 4 + 3]) * (1.f / 256.f);
  float ms   = (redB[row * 4] + redB[row * 4 + 1] + redB[row * 4 + 2] + redB[row * 4 + 3]) * (1.f / 256.f);
  float rstd = rsqrtf(ms - mean * mean + 1e-5f);
  for (int c = part * 64; c < part * 64 + 64; ++c) {
    float v = (ys[row * 260 + c] - mean) * rstd * gv[c] + bvv[c];
    t16[(r0 + row) * EDIM + c] = (_Float16)v;
  }
}

// ---------------- kernel 2: q/k/v GEMMs + attention + Wo + residual + LN -> h1 ----------------
__global__ __launch_bounds__(NTHR, 1)
void attn_kernel(const _Float16* __restrict__ t16, const unsigned int* __restrict__ pkQKV,
                 const unsigned int* __restrict__ pkWo,
                 const float* __restrict__ bq, const float* __restrict__ bk,
                 const float* __restrict__ bv, const float* __restrict__ bv_o,
                 const float* __restrict__ pb,
                 const float* __restrict__ g1, const float* __restrict__ b1,
                 _Float16* __restrict__ h1o) {
  __shared__ alignas(32) unsigned int Af[4 * 8 * 32 * 8];   // t fragments (32 KB)
  __shared__ _Float16 qkv[3][RPB * 264];                    // q,k,v tiles (~99 KB)
  __shared__ _Float16 ao[RPB * 264];                        // att_out tile (~33 KB)
  __shared__ float ys[RPB * 260];
  __shared__ float redA[256], redB[256];
  int tid = threadIdx.x, lane = tid & 31, wave = tid >> 5;
  size_t r0 = (size_t)blockIdx.x * RPB;

  const unsigned int* tsrc = (const unsigned int*)(const void*)(t16 + r0 * EDIM);
  for (int it = 0; it < 32; ++it) {
    int idx = it * NTHR + tid;
    int row = idx >> 7, pr = idx & 127;
    Af[fragAddrA(row, pr * 2)] = tsrc[row * 128 + pr];     // raw f16-pair copy
  }
  __syncthreads();

  const float* biases[3] = {bq, bk, bv};
  for (int job = wave; job < 48; job += 8) {    // 3 weights x 16 n-tiles
    int wsel = job >> 4, nt = job & 15;
    v8f acc[4] = {};
    gemm4<8>(Af, pkQKV + (size_t)wsel * 32768, 16, nt, lane, acc);
    int col = nt * 16 + (lane & 15);
    float bb = biases[wsel][col];
    int mbase = (lane >> 4) * 8;
#pragma unroll
    for (int rt = 0; rt < 4; ++rt)
#pragma unroll
      for (int i = 0; i < 8; ++i)
        qkv[wsel][(rt * 16 + mbase + i) * 264 + col] = (_Float16)(acc[rt][i] + bb);
  }
  __syncthreads();

  // attention: 8 groups x 8 heads x 8 queries = 512 jobs, seq len 8, d=32
  for (int p = 0; p < 2; ++p) {
    int idx = p * NTHR + tid;
    int g = idx >> 6, h = (idx >> 3) & 7, qi = idx & 7;
    int qrow = g * 8 + qi;
    const _Float16* Q = &qkv[0][qrow * 264 + h * 32];
    float sc[8]; float mx = -1e30f;
#pragma unroll
    for (int ki = 0; ki < 8; ++ki) {
      const _Float16* Kp = &qkv[1][(g * 8 + ki) * 264 + h * 32];
      float s = 0.f;
#pragma unroll
      for (int j = 0; j < 32; ++j) s += (float)Q[j] * (float)Kp[j];
      s = s * 0.17677669529663687f + pb[(r0 + g * 8 + ki) * 8 + h];
      sc[ki] = s; mx = fmaxf(mx, s);
    }
    float sum = 0.f;
#pragma unroll
    for (int ki = 0; ki < 8; ++ki) { sc[ki] = __expf(sc[ki] - mx); sum += sc[ki]; }
    float inv = 1.f / sum;
    for (int j = 0; j < 32; ++j) {
      float o = 0.f;
#pragma unroll
      for (int ki = 0; ki < 8; ++ki) o += sc[ki] * (float)qkv[2][(g * 8 + ki) * 264 + h * 32 + j];
      ao[qrow * 264 + h * 32 + j] = (_Float16)(o * inv);
    }
  }
  __syncthreads();

  for (int nt = wave; nt < 16; nt += 8) {       // att_out @ Wo + bo + t residual
    v8f acc[4] = {};
    gemm4_plain<8>(ao, 264, pkWo, 16, nt, lane, acc);
    int col = nt * 16 + (lane & 15);
    float bb = bv_o[col];
    int mbase = (lane >> 4) * 8;
#pragma unroll
    for (int rt = 0; rt < 4; ++rt)
#pragma unroll
      for (int i = 0; i < 8; ++i) {
        int row = rt * 16 + mbase + i;
        float tv = (float)t16[(r0 + row) * EDIM + col];
        ys[row * 260 + col] = acc[rt][i] + bb + tv;
      }
  }
  __syncthreads();

  int row = tid >> 2, part = tid & 3;
  float s = 0.f, s2 = 0.f;
  for (int c = part * 64; c < part * 64 + 64; ++c) { float v = ys[row * 260 + c]; s += v; s2 += v * v; }
  redA[tid] = s; redB[tid] = s2;
  __syncthreads();
  float mean = (redA[row * 4] + redA[row * 4 + 1] + redA[row * 4 + 2] + redA[row * 4 + 3]) * (1.f / 256.f);
  float ms   = (redB[row * 4] + redB[row * 4 + 1] + redB[row * 4 + 2] + redB[row * 4 + 3]) * (1.f / 256.f);
  float rstd = rsqrtf(ms - mean * mean + 1e-5f);
  for (int c = part * 64; c < part * 64 + 64; ++c) {
    float v = (ys[row * 260 + c] - mean) * rstd * g1[c] + b1[c];
    h1o[(r0 + row) * EDIM + c] = (_Float16)v;
  }
}

// ---------------- kernel 3: h2 = LN(h1 + silu(h1@Wff1)@Wff2) ----------------
__global__ __launch_bounds__(NTHR, 1)
void ffn_kernel(const _Float16* __restrict__ h1g, const unsigned int* __restrict__ pkF1,
                const unsigned int* __restrict__ pkF2,
                const float* __restrict__ gv, const float* __restrict__ bvv,
                float* __restrict__ out) {
  __shared__ alignas(32) unsigned int Af[4 * 8 * 32 * 8];   // h1 fragments (32 KB)
  __shared__ _Float16 us[RPB * 1032];                       // silu(h1@Wff1) tile (~129 KB)
  __shared__ float ys[RPB * 260];
  __shared__ float redA[256], redB[256];
  int tid = threadIdx.x, lane = tid & 31, wave = tid >> 5;
  size_t r0 = (size_t)blockIdx.x * RPB;

  const unsigned int* src = (const unsigned int*)(const void*)(h1g + r0 * EDIM);
  for (int it = 0; it < 32; ++it) {
    int idx = it * NTHR + tid;
    int row = idx >> 7, pr = idx & 127;
    Af[fragAddrA(row, pr * 2)] = src[row * 128 + pr];
  }
  __syncthreads();

  for (int nt = wave; nt < 64; nt += 8) {       // h1 @ Wff1 (N=1024), silu
    v8f acc[4] = {};
    gemm4<8>(Af, pkF1, 64, nt, lane, acc);
    int col = nt * 16 + (lane & 15);
    int mbase = (lane >> 4) * 8;
#pragma unroll
    for (int rt = 0; rt < 4; ++rt)
#pragma unroll
      for (int i = 0; i < 8; ++i)
        us[(rt * 16 + mbase + i) * 1032 + col] = (_Float16)silu_f(acc[rt][i]);
  }
  __syncthreads();

  for (int nt = wave; nt < 16; nt += 8) {       // u @ Wff2 (K=1024) + residual
    v8f acc[4] = {};
    gemm4_plain<32>(us, 1032, pkF2, 16, nt, lane, acc);
    int col = nt * 16 + (lane & 15);
    int mbase = (lane >> 4) * 8;
#pragma unroll
    for (int rt = 0; rt < 4; ++rt)
#pragma unroll
      for (int i = 0; i < 8; ++i) {
        int row = rt * 16 + mbase + i;
        ys[row * 260 + col] = acc[rt][i] + (float)h1g[(r0 + row) * EDIM + col];
      }
  }
  __syncthreads();

  int row = tid >> 2, part = tid & 3;
  float s = 0.f, s2 = 0.f;
  for (int c = part * 64; c < part * 64 + 64; ++c) { float v = ys[row * 260 + c]; s += v; s2 += v * v; }
  redA[tid] = s; redB[tid] = s2;
  __syncthreads();
  float mean = (redA[row * 4] + redA[row * 4 + 1] + redA[row * 4 + 2] + redA[row * 4 + 3]) * (1.f / 256.f);
  float ms   = (redB[row * 4] + redB[row * 4 + 1] + redB[row * 4 + 2] + redB[row * 4 + 3]) * (1.f / 256.f);
  float rstd = rsqrtf(ms - mean * mean + 1e-5f);
  for (int c = part * 64; c < part * 64 + 64; ++c) {
    float v = (ys[row * 260 + c] - mean) * rstd * gv[c] + bvv[c];
    out[(r0 + row) * EDIM + c] = v;
  }
}

extern "C" void kernel_launch(void* const* d_in, const int* in_sizes, int n_in,
                              void* d_out, int out_size, void* d_ws, size_t ws_size,
                              hipStream_t stream) {
  (void)in_sizes; (void)n_in; (void)out_size; (void)ws_size;
  const float* x     = (const float*)d_in[0];
  const float* pos1  = (const float*)d_in[1];
  const float* pos2  = (const float*)d_in[2];
  const float* W_in  = (const float*)d_in[3];
  const float* g_in  = (const float*)d_in[4];
  const float* b_in  = (const float*)d_in[5];
  const float* Wq    = (const float*)d_in[6];
  const float* bq    = (const float*)d_in[7];
  const float* Wk    = (const float*)d_in[8];
  const float* bk    = (const float*)d_in[9];
  const float* Wv    = (const float*)d_in[10];
  const float* bv    = (const float*)d_in[11];
  const float* Wo    = (const float*)d_in[12];
  const float* bo    = (const float*)d_in[13];
  const float* W_pos = (const float*)d_in[14];
  const float* W_bias= (const float*)d_in[15];
  const float* W_ff1 = (const float*)d_in[16];
  const float* W_ff2 = (const float*)d_in[17];
  const float* g1    = (const float*)d_in[18];
  const float* b1    = (const float*)d_in[19];
  const float* g2    = (const float*)d_in[20];
  const float* b2    = (const float*)d_in[21];

  // workspace layout (dword granularity for packed weights)
  unsigned int* pk = (unsigned int*)d_ws;
  const size_t PW256 = 32768;                         // dwords per packed 256x256
  unsigned int* pkWin = pk;
  unsigned int* pkWq  = pk + 1 * PW256;               // Wq, Wk, Wv contiguous
  unsigned int* pkWo_ = pk + 4 * PW256;
  unsigned int* pkF1  = pk + 5 * PW256;               // 131072 dwords
  unsigned int* pkF2  = pkF1 + 131072;                // 131072 dwords
  float*     pb   = (float*)(pkF2 + 131072);          // R_TOTAL*8 floats (4 MB)
  _Float16*  t16  = (_Float16*)(pb + (size_t)R_TOTAL * 8);   // 64 MB
  _Float16*  h116 = t16 + (size_t)R_TOTAL * EDIM;            // 64 MB

  pack_weight_kernel<<<128, 256, 0, stream>>>(W_in, pkWin, 256, 256);
  pack_weight_kernel<<<128, 256, 0, stream>>>(Wq, pk + 1 * PW256, 256, 256);
  pack_weight_kernel<<<128, 256, 0, stream>>>(Wk, pk + 2 * PW256, 256, 256);
  pack_weight_kernel<<<128, 256, 0, stream>>>(Wv, pk + 3 * PW256, 256, 256);
  pack_weight_kernel<<<128, 256, 0, stream>>>(Wo, pkWo_, 256, 256);
  pack_weight_kernel<<<512, 256, 0, stream>>>(W_ff1, pkF1, 256, 1024);
  pack_weight_kernel<<<512, 256, 0, stream>>>(W_ff2, pkF2, 1024, 256);
  posbias_kernel<<<R_TOTAL / 256, 256, 0, stream>>>(pos1, pos2, W_pos, W_bias, pb);

  inproj_ln_kernel<<<R_TOTAL / RPB, NTHR, 0, stream>>>(x, pkWin, g_in, b_in, t16);
  attn_kernel<<<R_TOTAL / RPB, NTHR, 0, stream>>>(t16, pkWq, pkWo_, bq, bk, bv, bo, pb, g1, b1, h116);
  ffn_kernel<<<R_TOTAL / RPB, NTHR, 0, stream>>>(h116, pkF1, pkF2, g2, b2, (float*)d_out);
}
